// MultiHeadSelfAttention_73890617360590
// MI455X (gfx1250) — compile-verified, matrix operand order checked by reading, not run on hardware
//
#include <hip/hip_runtime.h>

typedef _Float16 half16  __attribute__((ext_vector_type(16)));
typedef _Float16 half8v  __attribute__((ext_vector_type(8)));
typedef float    floatx8 __attribute__((ext_vector_type(8)));
typedef float    floatx4 __attribute__((ext_vector_type(4)));

#define T_SEQ  2048
#define DMODEL 1024
#define NH     16
#define DK     64
#define BATCH  4
#define MROWS  (BATCH * T_SEQ)   // 8192
#define QK_SCALE 0.125f          // 1/sqrt(64)

// ---------------------------------------------------------------------------
// Fragment loader: 16 contiguous halves per lane.
//   lanes 0-15 : row = row0 + lane,      halves [k0 .. k0+15]
//   lanes 16-31: row = row0 + (lane-16), halves [k0+16 .. k0+31]
// Used for both A-fragments (rows of left operand) and B-fragments (rows of
// the right operand's k-major source == columns of the B matrix).
// ---------------------------------------------------------------------------
__device__ __forceinline__ half16 load_frag(const _Float16* __restrict__ base,
                                            int ld, int row0, int k0) {
  int lane = threadIdx.x & 31;
  const _Float16* p = base + (size_t)(row0 + (lane & 15)) * ld + k0 + ((lane >> 4) << 4);
  return *(const half16*)p;
}

__device__ __forceinline__ floatx8 wmma_f16(half16 a, half16 b, floatx8 c) {
  return __builtin_amdgcn_wmma_f32_16x16x32_f16(false, a, false, b, (short)0, c,
                                                false, false);
}

// ---------------------------------------------------------------------------
// f32 -> f16 conversion (8 elements / thread, exact-divisible sizes)
// ---------------------------------------------------------------------------
__global__ void f32_to_f16_kernel(const float* __restrict__ in,
                                  _Float16* __restrict__ out, int n) {
  size_t i = ((size_t)blockIdx.x * blockDim.x + threadIdx.x) * 8;
  if (i + 8 > (size_t)n) return;
  floatx4 a = *(const floatx4*)(in + i);
  floatx4 b = *(const floatx4*)(in + i + 4);
  half8v h;
  h[0] = (_Float16)a[0]; h[1] = (_Float16)a[1];
  h[2] = (_Float16)a[2]; h[3] = (_Float16)a[3];
  h[4] = (_Float16)b[0]; h[5] = (_Float16)b[1];
  h[6] = (_Float16)b[2]; h[7] = (_Float16)b[3];
  *(half8v*)(out + i) = h;
}

// ---------------------------------------------------------------------------
// Projection GEMM: C[m,n] = sum_k A[m,k] * W[n,k]   (x @ W^T, torch Linear)
// One wave computes a 32(M) x 64(N) tile; K = 1024 in 32-wide steps.
// Per k-step: 2 A-frags + 4 B-frags (12 b128 loads) feed 8 WMMAs.
// mode 0: write f16 as (B,H,T,d_k)      (Q, K projections)
// mode 1: write f16 as (B,H,d_k,T)      (V projection, transposed for PV GEMM)
// mode 2: write f32 row-major (m,n)     (final output projection)
// ---------------------------------------------------------------------------
__global__ void proj_kernel(const _Float16* __restrict__ A,
                            const _Float16* __restrict__ W,
                            void* __restrict__ out, int mode) {
  int m0 = blockIdx.x * 32;
  int n0 = blockIdx.y * 64;
  int lane = threadIdx.x & 31;

  floatx8 c[2][4] = {};
  for (int kk = 0; kk < DMODEL; kk += 32) {
    half16 a0 = load_frag(A, DMODEL, m0,      kk);
    half16 a1 = load_frag(A, DMODEL, m0 + 16, kk);
    if (kk + 32 < DMODEL) {
      __builtin_prefetch(A + (size_t)m0 * DMODEL + kk + 32, 0, 0);
      __builtin_prefetch(A + (size_t)(m0 + 16) * DMODEL + kk + 32, 0, 0);
    }
#pragma unroll
    for (int t = 0; t < 4; ++t) {
      half16 b = load_frag(W, DMODEL, n0 + t * 16, kk);
      c[0][t] = wmma_f16(a0, b, c[0][t]);
      c[1][t] = wmma_f16(a1, b, c[1][t]);
    }
  }

  int grp = lane >> 4;
  int col = lane & 15;
#pragma unroll
  for (int mi = 0; mi < 2; ++mi) {
    for (int t = 0; t < 4; ++t) {
      for (int r = 0; r < 8; ++r) {
        int m = m0 + mi * 16 + r + (grp << 3);
        int n = n0 + t * 16 + col;
        float v = c[mi][t][r];
        if (mode == 2) {
          ((float*)out)[(size_t)m * DMODEL + n] = v;
        } else {
          int b  = m >> 11;              // m / T_SEQ
          int tt = m & (T_SEQ - 1);
          int h  = n >> 6;               // n / DK
          int d  = n & (DK - 1);
          size_t idx = (mode == 0)
              ? ((((size_t)(b * NH + h)) * T_SEQ + tt) * DK + d)      // (B,H,T,dk)
              : ((((size_t)(b * NH + h)) * DK + d) * T_SEQ + tt);     // (B,H,dk,T)
          ((_Float16*)out)[idx] = (_Float16)v;
        }
      }
    }
  }
}

// ---------------------------------------------------------------------------
// Causal flash attention: one wave per (b,h, 16-row q tile).
// Online softmax over 32-wide k blocks; O accumulated in 4 f32 C-fragments.
// P goes C-layout -> LDS -> A-layout for the PV WMMA.
// ---------------------------------------------------------------------------
__global__ void flash_attn_kernel(const _Float16* __restrict__ Q,
                                  const _Float16* __restrict__ K,
                                  const _Float16* __restrict__ Vt,
                                  _Float16* __restrict__ Out) {
  __shared__ __align__(32) _Float16 Pl[16 * 32];

  int qt   = blockIdx.x;          // q tile index, 0..127
  int bh   = blockIdx.y;          // fused (b,h), 0..63
  int lane = threadIdx.x & 31;
  int grp  = lane >> 4;
  int col  = lane & 15;

  const _Float16* Qb = Q  + (size_t)bh * T_SEQ * DK;
  const _Float16* Kb = K  + (size_t)bh * T_SEQ * DK;
  const _Float16* Vb = Vt + (size_t)bh * DK * T_SEQ;  // rows = d, cols = t

  half16 aq0 = load_frag(Qb, DK, qt * 16, 0);
  half16 aq1 = load_frag(Qb, DK, qt * 16, 32);

  float mrow[8], lrow[8];
  floatx8 o0{}, o1{}, o2{}, o3{};
  for (int r = 0; r < 8; ++r) { mrow[r] = -__builtin_inff(); lrow[r] = 0.f; }

  int jmax = (qt * 16 + 15) >> 5;          // last 32-wide k block (causal)
  for (int j = 0; j <= jmax; ++j) {
    int kbase = j * 32;

    // --- scores: S = Q @ K^T (contract over d = 64, two K32 steps) ---------
    floatx8 s0{}, s1{};
    s0 = wmma_f16(aq0, load_frag(Kb, DK, kbase,       0), s0);
    s0 = wmma_f16(aq1, load_frag(Kb, DK, kbase,      32), s0);
    s1 = wmma_f16(aq0, load_frag(Kb, DK, kbase + 16,  0), s1);
    s1 = wmma_f16(aq1, load_frag(Kb, DK, kbase + 16, 32), s1);

    // --- online softmax over this 32-wide block ----------------------------
    float p0[8], p1[8], alpha[8];
    for (int r = 0; r < 8; ++r) {
      int qrow = qt * 16 + r + (grp << 3);
      float v0 = (kbase + col      <= qrow) ? s0[r] * QK_SCALE : -__builtin_inff();
      float v1 = (kbase + 16 + col <= qrow) ? s1[r] * QK_SCALE : -__builtin_inff();
      float rm = fmaxf(v0, v1);
      for (int off = 1; off < 16; off <<= 1)
        rm = fmaxf(rm, __shfl_xor(rm, off, 32));   // stays in 16-lane half-group
      float mn = fmaxf(mrow[r], rm);
      alpha[r] = __expf(mrow[r] - mn);
      mrow[r]  = mn;
      p0[r] = __expf(v0 - mn);
      p1[r] = __expf(v1 - mn);
      float rs = p0[r] + p1[r];
      for (int off = 1; off < 16; off <<= 1)
        rs += __shfl_xor(rs, off, 32);
      lrow[r] = lrow[r] * alpha[r] + rs;
    }
    for (int r = 0; r < 8; ++r) {
      o0[r] *= alpha[r]; o1[r] *= alpha[r];
      o2[r] *= alpha[r]; o3[r] *= alpha[r];
    }

    // --- C-layout P -> LDS -> A-layout fragment ----------------------------
    for (int r = 0; r < 8; ++r) {
      int m = r + (grp << 3);
      Pl[m * 32 + col]      = (_Float16)p0[r];
      Pl[m * 32 + 16 + col] = (_Float16)p1[r];
    }
    asm volatile("s_wait_dscnt 0" ::: "memory");   // single wave: DS in-order
    half16 ap = *(const half16*)&Pl[(lane & 15) * 32 + (grp << 4)];

    // --- O += P @ V  (B-frag = rows of V^T, contiguous over k) -------------
    o0 = wmma_f16(ap, load_frag(Vb, T_SEQ,  0, kbase), o0);
    o1 = wmma_f16(ap, load_frag(Vb, T_SEQ, 16, kbase), o1);
    o2 = wmma_f16(ap, load_frag(Vb, T_SEQ, 32, kbase), o2);
    o3 = wmma_f16(ap, load_frag(Vb, T_SEQ, 48, kbase), o3);
  }

  // --- normalize and store as (B,T,H*dk) f16 for the output projection ----
  int b = bh >> 4, h = bh & 15;
  for (int r = 0; r < 8; ++r) {
    float inv = 1.f / lrow[r];
    int qrow = qt * 16 + r + (grp << 3);
    size_t rowoff = ((size_t)b * T_SEQ + qrow) * DMODEL + (size_t)h * DK;
    Out[rowoff +  0 + col] = (_Float16)(o0[r] * inv);
    Out[rowoff + 16 + col] = (_Float16)(o1[r] * inv);
    Out[rowoff + 32 + col] = (_Float16)(o2[r] * inv);
    Out[rowoff + 48 + col] = (_Float16)(o3[r] * inv);
  }
}

// ---------------------------------------------------------------------------
extern "C" void kernel_launch(void* const* d_in, const int* in_sizes, int n_in,
                              void* d_out, int out_size, void* d_ws, size_t ws_size,
                              hipStream_t stream) {
  const float* x  = (const float*)d_in[0];
  const float* Wq = (const float*)d_in[1];
  const float* Wk = (const float*)d_in[2];
  const float* Wv = (const float*)d_in[3];
  const float* Wo = (const float*)d_in[4];

  const size_t XN = (size_t)MROWS * DMODEL;    // 8 Mi halves
  const size_t WN = (size_t)DMODEL * DMODEL;   // 1 Mi halves

  _Float16* ws   = (_Float16*)d_ws;
  _Float16* x16  = ws;
  _Float16* wq16 = x16  + XN;
  _Float16* wk16 = wq16 + WN;
  _Float16* wv16 = wk16 + WN;
  _Float16* wo16 = wv16 + WN;
  _Float16* q16  = wo16 + WN;   // (B,H,T,dk)
  _Float16* k16  = q16  + XN;   // (B,H,T,dk)
  _Float16* vt16 = k16  + XN;   // (B,H,dk,T)
  _Float16* at16 = vt16 + XN;   // (B,T,D) attention output, f16

  const int cb = 256;
  f32_to_f16_kernel<<<(unsigned)(XN / 8 / cb), cb, 0, stream>>>(x,  x16,  (int)XN);
  f32_to_f16_kernel<<<(unsigned)(WN / 8 / cb), cb, 0, stream>>>(Wq, wq16, (int)WN);
  f32_to_f16_kernel<<<(unsigned)(WN / 8 / cb), cb, 0, stream>>>(Wk, wk16, (int)WN);
  f32_to_f16_kernel<<<(unsigned)(WN / 8 / cb), cb, 0, stream>>>(Wv, wv16, (int)WN);
  f32_to_f16_kernel<<<(unsigned)(WN / 8 / cb), cb, 0, stream>>>(Wo, wo16, (int)WN);

  dim3 pg(MROWS / 32, DMODEL / 64);            // 256 x 16 waves
  proj_kernel<<<pg, 32, 0, stream>>>(x16, wq16, q16,  0);
  proj_kernel<<<pg, 32, 0, stream>>>(x16, wk16, k16,  0);
  proj_kernel<<<pg, 32, 0, stream>>>(x16, wv16, vt16, 1);

  dim3 ag(T_SEQ / 16, BATCH * NH);             // 128 x 64 waves
  flash_attn_kernel<<<ag, 32, 0, stream>>>(q16, k16, vt16, at16);

  proj_kernel<<<pg, 32, 0, stream>>>(at16, wo16, d_out, 2);
}
